// GCN_51737176048479
// MI455X (gfx1250) — matrix operand order repaired and checked by reading, not compiled, not run on hardware
//
#include <hip/hip_runtime.h>
#include <hip/hip_bf16.h>

// ---------------------------------------------------------------------------
// GCN on MI455X (gfx1250): WMMA f16 GEMMs + L2-resident edge scatter-gather.
// ---------------------------------------------------------------------------

typedef __attribute__((ext_vector_type(16))) _Float16 v16h;
typedef __attribute__((ext_vector_type(8)))  _Float16 v8h;
typedef __attribute__((ext_vector_type(4)))  _Float16 v4h;
typedef __attribute__((ext_vector_type(8)))  float    v8f;

#define N_NODES 100000
#define N_EDGES 3200000
#define F_IN    128
#define H1      64
#define H2      32
#define N_CLS   32

// ---------------------------------------------------------------------------
// 16x32 f16 A-fragment (ISA 7.12.2 16-bit A layout).
// lane 0-15  : row = lane,    K = kb+0..7  (halfs 0-7), kb+16..23 (halfs 8-15)
// lane 16-31 : row = lane-16, K = kb+8..15 (halfs 0-7), kb+24..31 (halfs 8-15)
// ---------------------------------------------------------------------------
__device__ __forceinline__ v16h load_a_frag(const _Float16* __restrict__ A,
                                            int lda, int m0, int kb, int lane) {
    const _Float16* p = A + (size_t)(m0 + (lane & 15)) * lda + kb + ((lane >> 4) << 3);
    v8h lo = *(const v8h*)(p);
    v8h hi = *(const v8h*)(p + 16);
    v16h a;
#pragma unroll
    for (int i = 0; i < 8; ++i) { a[i] = lo[i]; a[i + 8] = hi[i]; }
    return a;
}

// B-fragment: weights pre-transposed to Wt[N][K]; lane holds column n = lane&15,
// K = kb + (lane>>4)*8 + 0..7 (halfs 0-7), +16 (halfs 8-15) -> matches A layout.
__device__ __forceinline__ v16h load_b_frag(const _Float16* __restrict__ Wt,
                                            int K, int n0, int kb, int lane) {
    const _Float16* p = Wt + (size_t)(n0 + (lane & 15)) * K + kb + ((lane >> 4) << 3);
    v8h lo = *(const v8h*)(p);
    v8h hi = *(const v8h*)(p + 16);
    v16h b;
#pragma unroll
    for (int i = 0; i < 8; ++i) { b[i] = lo[i]; b[i + 8] = hi[i]; }
    return b;
}

// ---------------------------------------------------------------------------
// Degree / norm kernels
// ---------------------------------------------------------------------------
__global__ void k_deg_init(float* __restrict__ deg, int n) {
    int i = blockIdx.x * blockDim.x + threadIdx.x;
    if (i < n) deg[i] = 1.0f;                       // self-loop
}

__global__ void k_deg_edges(const int* __restrict__ ei, float* __restrict__ deg, int ne) {
    int e = blockIdx.x * blockDim.x + threadIdx.x;
    if (e < ne) unsafeAtomicAdd(&deg[ei[ne + e]], 1.0f);   // dst = row 1
}

__global__ void k_dinv(const float* __restrict__ deg, float* __restrict__ dinv, int n) {
    int i = blockIdx.x * blockDim.x + threadIdx.x;
    if (i < n) dinv[i] = rsqrtf(deg[i]);
}

// ---------------------------------------------------------------------------
// f32 -> f16 cast (optionally fused relu), 4 elems/thread
// ---------------------------------------------------------------------------
template<bool RELU>
__global__ void k_cast_f16(const float* __restrict__ src, _Float16* __restrict__ dst, int n4) {
    int t = blockIdx.x * blockDim.x + threadIdx.x;
    if (t >= n4) return;
    float4 v = ((const float4*)src)[t];
    if (RELU) {
        v.x = fmaxf(v.x, 0.f); v.y = fmaxf(v.y, 0.f);
        v.z = fmaxf(v.z, 0.f); v.w = fmaxf(v.w, 0.f);
    }
    v4h o;
    o[0] = (_Float16)v.x; o[1] = (_Float16)v.y;
    o[2] = (_Float16)v.z; o[3] = (_Float16)v.w;
    ((v4h*)dst)[t] = o;
}

// Transpose + cast all weights once per call (tiny).
__global__ void k_prep_weights(const float* __restrict__ W1, const float* __restrict__ W2,
                               const float* __restrict__ Wf, const float* __restrict__ Wo,
                               _Float16* __restrict__ W1t, _Float16* __restrict__ W2t,
                               _Float16* __restrict__ Wft, _Float16* __restrict__ Wot) {
    for (int i = threadIdx.x; i < F_IN * H1; i += blockDim.x) {
        int k = i / H1, n = i % H1;
        W1t[n * F_IN + k] = (_Float16)W1[i];
    }
    for (int i = threadIdx.x; i < H1 * H2; i += blockDim.x) {
        int k = i / H2, n = i % H2;
        W2t[n * H1 + k] = (_Float16)W2[i];
    }
    for (int i = threadIdx.x; i < H2 * H2; i += blockDim.x) {
        int k = i / H2, n = i % H2;
        Wft[n * H2 + k] = (_Float16)Wf[i];
        Wot[n * H2 + k] = (_Float16)Wo[i];
    }
}

// ---------------------------------------------------------------------------
// WMMA GEMM: C[M x (NT*16)] = A[M x K] (f16) * Wt^T (f16), f32 accumulate.
// One 16x16 output tile per wave. M must be a multiple of 16 (100000 = 6250*16).
// ---------------------------------------------------------------------------
template<int K, int NT>
__global__ void k_gemm_wmma(const _Float16* __restrict__ A, const _Float16* __restrict__ Wt,
                            float* __restrict__ C, int mtiles) {
    const int lane = threadIdx.x & 31;
    const int wave = threadIdx.x >> 5;
    const int tile = blockIdx.x * (blockDim.x >> 5) + wave;
    if (tile >= mtiles * NT) return;
    const int m0 = (tile / NT) * 16;
    const int n0 = (tile % NT) * 16;

    v8f c = {};
#pragma unroll
    for (int kb = 0; kb < K; kb += 32) {
        v16h a = load_a_frag(A, K, m0, kb, lane);
        v16h b = load_b_frag(Wt, K, n0, kb, lane);
        c = __builtin_amdgcn_wmma_f32_16x16x32_f16(false, a, false, b,
                                                   (short)0, c, false, false);
    }
    const int col = n0 + (lane & 15);
    const int r0  = m0 + ((lane >> 4) << 3);
#pragma unroll
    for (int v = 0; v < 8; ++v)
        C[(size_t)(r0 + v) * (NT * 16) + col] = c[v];
}

// ---------------------------------------------------------------------------
// GCN aggregation.
// init: agg[i][f] = h[i][f]*dinv[i]^2 + b[f]     (self-loop + bias folded in)
// edge: agg[dst][f] += h[src][f]*dinv[src]*dinv[dst]
// ---------------------------------------------------------------------------
template<int F>   // 4 features per thread
__global__ void k_conv_init(const float* __restrict__ h, const float* __restrict__ dinv,
                            const float* __restrict__ b, float* __restrict__ agg, int n) {
    int t = blockIdx.x * blockDim.x + threadIdx.x;          // n * F/4 threads
    if (t >= n * (F / 4)) return;
    int i  = t / (F / 4);
    int f4 = (t % (F / 4)) * 4;
    float w = dinv[i];
    float ww = w * w;
    float4 hv = ((const float4*)h)[t];
    float4 o;
    o.x = hv.x * ww + b[f4 + 0];
    o.y = hv.y * ww + b[f4 + 1];
    o.z = hv.z * ww + b[f4 + 2];
    o.w = hv.w * ww + b[f4 + 3];
    ((float4*)agg)[t] = o;
}

// One WAVE per edge. Edge id is wave-uniform -> force it into an SGPR with
// readfirstlane so the index / norm loads become s_load_b32 (constant cache),
// keeping the vector memory pipe free for the gather + scatter atomics.
template<int F>
__global__ void k_conv_edges(const float* __restrict__ h, const float* __restrict__ dinv,
                             const int* __restrict__ ei, float* __restrict__ agg, int ne) {
    constexpr int V = F / 32;                                // features per lane
    const int lane = threadIdx.x & 31;
    int e = (int)(blockIdx.x * (blockDim.x >> 5) + (threadIdx.x >> 5));
    e = __builtin_amdgcn_readfirstlane(e);
    if (e >= ne) return;
    const int s = ei[e];                                     // scalar loads
    const int d = ei[ne + e];
    const float w = dinv[s] * dinv[d];
    const float* __restrict__ hp = h   + (size_t)s * F + lane * V;
    float* __restrict__       ap = agg + (size_t)d * F + lane * V;
    if (V == 2) {
        float2 hv = *(const float2*)hp;
        unsafeAtomicAdd(ap + 0, hv.x * w);
        unsafeAtomicAdd(ap + 1, hv.y * w);
    } else {
        unsafeAtomicAdd(ap, hp[0] * w);
    }
}

// ---------------------------------------------------------------------------
// Fused MLP head: out = relu(agg2 @ Wf + bf) @ Wo + bo.
// One 16-row tile per wave; intermediate 16x32 tile bounced through LDS to
// convert the f32 D layout into the f16 A layout for the second WMMA.
// blockDim = 320 (10 waves), 625 blocks -> exactly 6250 tiles.
// ---------------------------------------------------------------------------
__global__ void k_mlp_fused(const _Float16* __restrict__ A,      // agg2 f16 [N][32]
                            const _Float16* __restrict__ Wft, const float* __restrict__ bf,
                            const _Float16* __restrict__ Wot, const float* __restrict__ bo,
                            float* __restrict__ out) {
    __shared__ _Float16 stage[10][16 * 32];
    const int lane = threadIdx.x & 31;
    const int wave = threadIdx.x >> 5;
    const int m0   = (blockIdx.x * 10 + wave) * 16;
    const int ln   = lane & 15;
    const int lh   = lane >> 4;

    // ---- GEMM A: t = relu(agg2 @ Wf + bf) ----
    v16h a  = load_a_frag(A, H2, m0, 0, lane);
    v16h b0 = load_b_frag(Wft, H2, 0,  0, lane);
    v16h b1 = load_b_frag(Wft, H2, 16, 0, lane);
    v8f c0, c1;
    {
        float f0 = bf[ln], f1 = bf[16 + ln];
#pragma unroll
        for (int v = 0; v < 8; ++v) { c0[v] = f0; c1[v] = f1; }
    }
    c0 = __builtin_amdgcn_wmma_f32_16x16x32_f16(false, a, false, b0, (short)0, c0, false, false);
    c1 = __builtin_amdgcn_wmma_f32_16x16x32_f16(false, a, false, b1, (short)0, c1, false, false);

    // relu + convert + stage in A layout (row-major 16x32 f16 tile in LDS)
    _Float16* s = stage[wave];
#pragma unroll
    for (int v = 0; v < 8; ++v) {
        int row = (lh << 3) + v;
        s[row * 32 + ln]      = (_Float16)fmaxf(c0[v], 0.f);
        s[row * 32 + 16 + ln] = (_Float16)fmaxf(c1[v], 0.f);
    }
    __syncthreads();

    // ---- GEMM B: out = t @ Wo + bo ----
    v16h t;
    {
        const _Float16* p = s + ln * 32 + (lh << 3);
        v8h lo = *(const v8h*)(p);
        v8h hi = *(const v8h*)(p + 16);
#pragma unroll
        for (int i = 0; i < 8; ++i) { t[i] = lo[i]; t[i + 8] = hi[i]; }
    }
    v16h w0 = load_b_frag(Wot, H2, 0,  0, lane);
    v16h w1 = load_b_frag(Wot, H2, 16, 0, lane);
    v8f d0, d1;
    {
        float f0 = bo[ln], f1 = bo[16 + ln];
#pragma unroll
        for (int v = 0; v < 8; ++v) { d0[v] = f0; d1[v] = f1; }
    }
    d0 = __builtin_amdgcn_wmma_f32_16x16x32_f16(false, t, false, w0, (short)0, d0, false, false);
    d1 = __builtin_amdgcn_wmma_f32_16x16x32_f16(false, t, false, w1, (short)0, d1, false, false);

#pragma unroll
    for (int v = 0; v < 8; ++v) {
        int row = m0 + (lh << 3) + v;
        out[(size_t)row * N_CLS + ln]      = d0[v];
        out[(size_t)row * N_CLS + 16 + ln] = d1[v];
    }
}

// ---------------------------------------------------------------------------
// Host driver
// ---------------------------------------------------------------------------
extern "C" void kernel_launch(void* const* d_in, const int* in_sizes, int n_in,
                              void* d_out, int out_size, void* d_ws, size_t ws_size,
                              hipStream_t stream) {
    const float* x  = (const float*)d_in[0];
    const int*   ei = (const int*)d_in[1];
    const float* W1 = (const float*)d_in[2]; const float* b1 = (const float*)d_in[3];
    const float* W2 = (const float*)d_in[4]; const float* b2 = (const float*)d_in[5];
    const float* Wf = (const float*)d_in[6]; const float* bf = (const float*)d_in[7];
    const float* Wo = (const float*)d_in[8]; const float* bo = (const float*)d_in[9];
    float* out = (float*)d_out;

    constexpr size_t MB = 1ull << 20;
    char* ws = (char*)d_ws;
    // lifetime-aliased scratch layout (peak ~95 MB)
    float*    deg    = (float*)(ws + 0);
    float*    dinv   = (float*)(ws + (MB / 2));
    _Float16* W1t    = (_Float16*)(ws + 1 * MB);
    _Float16* W2t    = (_Float16*)(ws + 1 * MB + 64 * 1024);
    _Float16* Wft    = (_Float16*)(ws + 1 * MB + 96 * 1024);
    _Float16* Wot    = (_Float16*)(ws + 1 * MB + 128 * 1024);
    _Float16* xh     = (_Float16*)(ws + 4 * MB);    // 25.6 MB, dead after GEMM1
    float*    h2     = (float*)   (ws + 4 * MB);    // alias: 12.8 MB
    float*    agg2   = (float*)   (ws + 17 * MB);   // 12.8 MB
    float*    h1     = (float*)   (ws + 30 * MB);   // 25.6 MB, dead after conv1 edges
    _Float16* agg2h  = (_Float16*)(ws + 30 * MB);   // alias: 6.4 MB
    float*    agg1   = (float*)   (ws + 56 * MB);   // 25.6 MB
    _Float16* r1h    = (_Float16*)(ws + 82 * MB);   // 12.8 MB

    const int TB = 256;                     // 8 waves per block

    // 1) symmetric normalization  D^-1/2 (self-loops included)
    k_deg_init <<<(N_NODES + TB - 1) / TB, TB, 0, stream>>>(deg, N_NODES);
    k_deg_edges<<<N_EDGES / TB,            TB, 0, stream>>>(ei, deg, N_EDGES);
    k_dinv     <<<(N_NODES + TB - 1) / TB, TB, 0, stream>>>(deg, dinv, N_NODES);

    // 2) precision prep
    k_cast_f16<false><<<(N_NODES * F_IN / 4) / TB, TB, 0, stream>>>(x, xh, N_NODES * F_IN / 4);
    k_prep_weights<<<1, 256, 0, stream>>>(W1, W2, Wf, Wo, W1t, W2t, Wft, Wot);

    // 3) conv1: h1 = x @ W1 ; agg1 = scatter(norm * h1[src]) + b1 ; relu -> f16
    k_gemm_wmma<F_IN, H1 / 16><<<(N_NODES / 16) * (H1 / 16) / 8, 256, 0, stream>>>(xh, W1t, h1, N_NODES / 16);
    k_conv_init <H1><<<(size_t)N_NODES * (H1 / 4) / TB, TB, 0, stream>>>(h1, dinv, b1, agg1, N_NODES);
    k_conv_edges<H1><<<N_EDGES / 8, TB, 0, stream>>>(h1, dinv, ei, agg1, N_EDGES);   // 1 wave/edge
    k_cast_f16<true><<<(N_NODES * H1 / 4) / TB, TB, 0, stream>>>(agg1, r1h, N_NODES * H1 / 4);

    // 4) conv2: h2 = relu1 @ W2 ; agg2 = scatter(norm * h2[src]) + b2
    {
        int tiles = (N_NODES / 16) * (H2 / 16);
        k_gemm_wmma<H1, H2 / 16><<<(tiles + 7) / 8, 256, 0, stream>>>(r1h, W2t, h2, N_NODES / 16);
    }
    k_conv_init <H2><<<(size_t)N_NODES * (H2 / 4) / TB, TB, 0, stream>>>(h2, dinv, b2, agg2, N_NODES);
    k_conv_edges<H2><<<N_EDGES / 8, TB, 0, stream>>>(h2, dinv, ei, agg2, N_EDGES);   // 1 wave/edge
    k_cast_f16<false><<<(N_NODES * H2 / 4) / TB, TB, 0, stream>>>(agg2, agg2h, N_NODES * H2 / 4);

    // 5) fused MLP head -> d_out
    k_mlp_fused<<<N_NODES / 16 / 10, 320, 0, stream>>>(agg2h, Wft, bf, Wot, bo, out);
}
